// Transformer_38259568673461
// MI455X (gfx1250) — matrix operand order repaired and checked by reading, not compiled, not run on hardware
//
#include <hip/hip_runtime.h>
#include <hip/hip_bf16.h>

// ---------------------------------------------------------------------------
// Transformer forward for MI455X (gfx1250, wave32, WMMA + Tensor Data Mover).
// GEMMs consume bf16 A/B tiles DMA'd into double-buffered LDS by TDM
// (tensor_load_to_lds, TENSORcnt), accumulate f32 via v_wmma_f32_16x16x32_bf16.
// Weights pre-transposed+converted to bf16 [N,K] once; activations written as
// bf16 by each GEMM epilogue (residual stream stays f32).
// ---------------------------------------------------------------------------

typedef __bf16 bf16_t;
typedef __attribute__((ext_vector_type(16))) __bf16 v16bf;
typedef __attribute__((ext_vector_type(8)))  __bf16 v8bf;
typedef __attribute__((ext_vector_type(8)))  float  v8f;
typedef __attribute__((ext_vector_type(4)))  unsigned int u32x4;
typedef __attribute__((ext_vector_type(8)))  int i32x8;
typedef __attribute__((ext_vector_type(4)))  int i32x4;

static constexpr int D_ = 1024;
static constexpr int H_ = 4096;
static constexpr int V_ = 32000;
static constexpr int L_ = 4;
static constexpr int B_ = 2;
static constexpr int S_ = 2048;
static constexpr float SCALE_ = 0.03125f;   // 1/sqrt(1024)
static constexpr float EPS_ = 1e-5f;

__device__ __forceinline__ bf16_t f2bf(float f) {
  unsigned u = __builtin_bit_cast(unsigned, f);
  u += 0x7fffu + ((u >> 16) & 1u);           // round-to-nearest-even
  unsigned short h = (unsigned short)(u >> 16);
  return __builtin_bit_cast(bf16_t, h);
}

// ---- staging-path selection ------------------------------------------------
#if __has_builtin(__builtin_amdgcn_tensor_load_to_lds) && \
    __has_builtin(__builtin_amdgcn_s_wait_tensorcnt)
#define USE_TDM 1
#else
#define USE_TDM 0
#endif
#if !USE_TDM && __has_builtin(__builtin_amdgcn_load_to_lds)
#define USE_ASYNC_LDS 1
#else
#define USE_ASYNC_LDS 0
#endif

template <typename T>
__device__ __forceinline__ __attribute__((address_space(3))) T* as3(T* p) {
  return (__attribute__((address_space(3))) T*)(unsigned)(unsigned long long)p;
}
__device__ __forceinline__ unsigned lds_off(const void* p) {
  return (unsigned)(unsigned long long)p;     // generic->LDS-offset truncation
}

// ---------------------------------------------------------------------------
// GEMM tile geometry
// ---------------------------------------------------------------------------
#define BM 128
#define BN 128
#define BK 64
#define LSTR 72        // padded LDS row stride in bf16 elems (144B rows)
// TDM LDS padding: after every 128B (32 DW, code 4) insert 16B (4 DW, code 3)
#define PAD_IVL_CODE 4
#define PAD_AMT_CODE 3

#if USE_TDM
// DMA one [128 x BK] bf16 tile (row stride = strideElems) into padded LDS.
// All descriptor fields are block-uniform -> SGPR descriptor groups.
__device__ __forceinline__ void tdm_load_tile(const bf16_t* gsrc, bf16_t* ldst,
                                              int strideElems) {
  const unsigned long long ga = (unsigned long long)gsrc;
  u32x4 g0;
  g0[0] = 1u;                                       // count=1 (valid, user)
  g0[1] = lds_off(ldst);                            // lds_addr (bytes)
  g0[2] = (unsigned)ga;                             // global_addr[31:0]
  g0[3] = ((unsigned)(ga >> 32) & 0x01FFFFFFu)      // global_addr[56:32]
          | 0x80000000u;                            // type=2 ("image")
  i32x8 g1;
  g1[0] = (int)((1u << 16)                          // data_size = 2 bytes
              | (1u << 20)                          // pad_enable
              | ((unsigned)PAD_IVL_CODE << 22)
              | ((unsigned)PAD_AMT_CODE << 25));
  g1[1] = (int)((unsigned)BK << 16);                // tensor_dim0 = BK
  g1[2] = (int)(128u << 16);                        // tensor_dim1 = 128
  g1[3] = (int)((unsigned)BK << 16);                // tile_dim0 = BK
  g1[4] = 128;                                      // tile_dim1 = 128
  g1[5] = strideElems;                              // tensor_dim0_stride
  g1[6] = 0; g1[7] = 0;
  const i32x4 z4 = {0, 0, 0, 0};
  const i32x8 z8 = {0, 0, 0, 0, 0, 0, 0, 0};
  // amdgpu-toolchain (clang-23) 6-arg form: (g0, g1, g2, g3, g4, cpol)
  __builtin_amdgcn_tensor_load_to_lds(g0, g1, z4, z4, z8, 0);
}
#endif

// ---------------------------------------------------------------------------
// bf16 WMMA GEMM, B pre-transposed to [N,K]:
//   Cf/Cb[z] = epi(alpha * A[z] @ B[z]^T (+bias) (+resid), relu?)
// Tile 128x128x64; 256 thr = 8 waves (2M x 4N); wave tile 64x32 = 4x2 accums.
// flags: 1=relu 2=resid(f32) 4=bias 8=causal-skip-upper 16=causal-K-limit
//        32=store f32 64=store bf16
// ---------------------------------------------------------------------------
__global__ __launch_bounds__(256)
void gemm_bf16_wmma(const bf16_t* __restrict__ A, const bf16_t* __restrict__ Bm,
                    const float* __restrict__ bias, const float* __restrict__ resid,
                    float* __restrict__ Cf, bf16_t* __restrict__ Cb,
                    int M, int N, int K,
                    long long sA, long long sB, long long sC,
                    float alpha, int flags)
{
  const int m0 = blockIdx.y * BM;
  const int n0 = blockIdx.x * BN;
  if ((flags & 8) && n0 > m0 + (BM - 1)) return;   // fully-masked causal tile

  const long long z = blockIdx.z;
  A  += z * sA;
  Bm += z * sB;
  const float*  Rp  = resid + z * sC;
  float*        Cfp = Cf + z * sC;
  bf16_t*       Cbp = Cb + z * sC;

  const int tid  = threadIdx.x;
  const int lane = tid & 31;
  const int wid  = tid >> 5;
  const int wm   = wid >> 2;     // 0..1
  const int wn   = wid & 3;      // 0..3

  __shared__ __align__(16) bf16_t As[2][BM * LSTR];   // [m][k] (padded rows)
  __shared__ __align__(16) bf16_t Bs[2][BN * LSTR];   // [n][k]

  v8f acc[4][2];
  #pragma unroll
  for (int i = 0; i < 4; ++i)
    #pragma unroll
    for (int j = 0; j < 2; ++j) acc[i][j] = v8f{};

#if !USE_TDM
  // manual staging: thread -> (row, 64B half-row)
  const int srow = tid >> 1;              // 0..127
  const int scol = (tid & 1) * 32;        // elem offset 0 / 32
  const long long agoff = (long long)(m0 + srow) * K + scol;
  const long long bgoff = (long long)(n0 + srow) * K + scol;
#endif

#if USE_TDM
  auto issue = [&](int k0, int buf) {
    if (wid == 0) {
      tdm_load_tile(A  + (long long)m0 * K + k0, &As[buf][0], K);
      tdm_load_tile(Bm + (long long)n0 * K + k0, &Bs[buf][0], K);
    }
  };
#elif USE_ASYNC_LDS
  auto issue = [&](int k0, int buf) {
    const bf16_t* ga = A  + agoff + k0;
    const bf16_t* gb = Bm + bgoff + k0;
    bf16_t* la = &As[buf][srow * LSTR + scol];
    bf16_t* lb = &Bs[buf][srow * LSTR + scol];
    #pragma unroll
    for (int c = 0; c < 4; ++c) {
      __builtin_amdgcn_load_to_lds((void*)(ga + 8 * c), as3((void*)(la + 8 * c)), 16, 0, 0);
      __builtin_amdgcn_load_to_lds((void*)(gb + 8 * c), as3((void*)(lb + 8 * c)), 16, 0, 0);
    }
  };
#else
  uint4 ra[4], rb[4];
  auto fetch = [&](int k0) {
    #pragma unroll
    for (int c = 0; c < 4; ++c) {
      ra[c] = *reinterpret_cast<const uint4*>(A  + agoff + k0 + 8 * c);
      rb[c] = *reinterpret_cast<const uint4*>(Bm + bgoff + k0 + 8 * c);
    }
  };
  auto commit = [&](int buf) {
    #pragma unroll
    for (int c = 0; c < 4; ++c) {
      *reinterpret_cast<uint4*>(&As[buf][srow * LSTR + scol + 8 * c]) = ra[c];
      *reinterpret_cast<uint4*>(&Bs[buf][srow * LSTR + scol + 8 * c]) = rb[c];
    }
  };
#endif

  auto compute = [&](int buf, int sub) {   // one 16x16x32 k-substep (8 WMMAs)
    const int kofs = sub * 32;
    v16bf afrag[4];
    const int akb = (lane >> 4) * 8 + kofs;
    #pragma unroll
    for (int im = 0; im < 4; ++im) {
      const int mrow = wm * 64 + im * 16 + (lane & 15);
      const v8bf lo = *reinterpret_cast<const v8bf*>(&As[buf][mrow * LSTR + akb]);
      const v8bf hi = *reinterpret_cast<const v8bf*>(&As[buf][mrow * LSTR + akb + 16]);
      v16bf a;
      #pragma unroll
      for (int e = 0; e < 8; ++e) { a[e] = lo[e]; a[e + 8] = hi[e]; }
      afrag[im] = a;
    }
    v16bf bfrag[2];
    const int bkb = (lane >> 4) * 8 + kofs;
    #pragma unroll
    for (int jn = 0; jn < 2; ++jn) {
      const int ncol = wn * 32 + jn * 16 + (lane & 15);
      const v8bf lo = *reinterpret_cast<const v8bf*>(&Bs[buf][ncol * LSTR + bkb]);
      const v8bf hi = *reinterpret_cast<const v8bf*>(&Bs[buf][ncol * LSTR + bkb + 16]);
      v16bf b;
      #pragma unroll
      for (int e = 0; e < 8; ++e) { b[e] = lo[e]; b[e + 8] = hi[e]; }
      bfrag[jn] = b;
    }
    #pragma unroll
    for (int im = 0; im < 4; ++im)
      #pragma unroll
      for (int jn = 0; jn < 2; ++jn)
        acc[im][jn] = __builtin_amdgcn_wmma_f32_16x16x32_bf16(
            false, afrag[im], false, bfrag[jn],
            (short)0, acc[im][jn], false, false);
  };

  const int Kend = (flags & 16) ? ((m0 + BM) < K ? (m0 + BM) : K) : K;
  const int nk = Kend / BK;

#if USE_TDM
  issue(0, 0);
  for (int kt = 0; kt < nk; ++kt) {
    __builtin_amdgcn_s_wait_tensorcnt(0);   // wave0: stage kt landed (others: 0)
    __syncthreads();                        // broadcast; prior reads finished
    if (kt + 1 < nk) issue((kt + 1) * BK, (kt + 1) & 1);  // DMA overlaps WMMAs
    compute(kt & 1, 0);
    compute(kt & 1, 1);
  }
#elif USE_ASYNC_LDS
  issue(0, 0);
  for (int kt = 0; kt < nk; ++kt) {
#if __has_builtin(__builtin_amdgcn_s_wait_asynccnt)
    __builtin_amdgcn_s_wait_asynccnt(0);
#else
    asm volatile("s_wait_asynccnt 0x0" ::: "memory");
#endif
    __syncthreads();
    if (kt + 1 < nk) issue((kt + 1) * BK, (kt + 1) & 1);
    compute(kt & 1, 0);
    compute(kt & 1, 1);
  }
#else
  fetch(0); commit(0);
  for (int kt = 0; kt < nk; ++kt) {
    __syncthreads();
    if (kt + 1 < nk) fetch((kt + 1) * BK);
    compute(kt & 1, 0);
    compute(kt & 1, 1);
    if (kt + 1 < nk) commit((kt + 1) & 1);
  }
#endif

  // epilogue: C/D layout -> lane n = lane%16, vgpr v -> m = (lane/16)*8+v
  const bool do_relu = flags & 1, has_res = flags & 2, has_bias = flags & 4;
  const bool st_f = flags & 32, st_b = flags & 64;
  #pragma unroll
  for (int im = 0; im < 4; ++im) {
    #pragma unroll
    for (int jn = 0; jn < 2; ++jn) {
      const int nc = n0 + wn * 32 + jn * 16 + (lane & 15);
      const float bv = has_bias ? bias[nc] : 0.f;
      #pragma unroll
      for (int v = 0; v < 8; ++v) {
        const int mr = m0 + wm * 64 + im * 16 + (lane >> 4) * 8 + v;
        float val = acc[im][jn][v] * alpha + bv;
        if (has_res) val += Rp[(long long)mr * N + nc];
        if (do_relu) val = val > 0.f ? val : 0.f;
        if (st_f) Cfp[(long long)mr * N + nc] = val;
        if (st_b) Cbp[(long long)mr * N + nc] = f2bf(val);
      }
    }
  }
}

// ---------------------------------------------------------------------------
// f32 [R,C] -> bf16 [C,R] tiled transpose+convert (weights)
// ---------------------------------------------------------------------------
__global__ __launch_bounds__(256)
void tcvt_f32_bf16(const float* __restrict__ src, bf16_t* __restrict__ dst,
                   int R, int C)
{
  __shared__ float t[32][33];
  const int tx = threadIdx.x & 31, ty = threadIdx.x >> 5;
  const int r0 = blockIdx.y * 32, c0 = blockIdx.x * 32;
  #pragma unroll
  for (int i = 0; i < 4; ++i)
    t[ty + 8 * i][tx] = src[(long long)(r0 + ty + 8 * i) * C + (c0 + tx)];
  __syncthreads();
  #pragma unroll
  for (int i = 0; i < 4; ++i)
    dst[(long long)(c0 + ty + 8 * i) * R + (r0 + tx)] = f2bf(t[tx][ty + 8 * i]);
}

// bf16 [R,C] -> bf16 [C,R], batched over z (V -> V^T)
__global__ __launch_bounds__(256)
void transpose_bf16(const bf16_t* __restrict__ src, bf16_t* __restrict__ dst,
                    int R, int C, long long sS, long long sD)
{
  src += (long long)blockIdx.z * sS;
  dst += (long long)blockIdx.z * sD;
  __shared__ bf16_t t[32][33];
  const int tx = threadIdx.x & 31, ty = threadIdx.x >> 5;
  const int r0 = blockIdx.y * 32, c0 = blockIdx.x * 32;
  #pragma unroll
  for (int i = 0; i < 4; ++i)
    t[ty + 8 * i][tx] = src[(long long)(r0 + ty + 8 * i) * C + (c0 + tx)];
  __syncthreads();
  #pragma unroll
  for (int i = 0; i < 4; ++i)
    dst[(long long)(c0 + ty + 8 * i) * R + (r0 + tx)] = t[tx][ty + 8 * i];
}

// ---------------------------------------------------------------------------
__global__ __launch_bounds__(256)
void embed_kernel(const int* __restrict__ tokens, const float* __restrict__ tok_emb,
                  const float* __restrict__ pos_emb,
                  float* __restrict__ x, bf16_t* __restrict__ xb)
{
  const int row = blockIdx.x;
  const int s   = row & (S_ - 1);
  const int tok = tokens[row];
  const float* te = tok_emb + (long long)tok * D_;
  const float* pe = pos_emb + (long long)s   * D_;
  float*  of = x  + (long long)row * D_;
  bf16_t* ob = xb + (long long)row * D_;
  for (int i = threadIdx.x; i < D_; i += 256) {
    const float v = te[i] + pe[i];
    of[i] = v; ob[i] = f2bf(v);
  }
}

// ---------------------------------------------------------------------------
__global__ __launch_bounds__(256)
void softmax_causal(const float* __restrict__ scores, bf16_t* __restrict__ probs)
{
  const int row = blockIdx.x;           // b*S + q
  const int q   = row & (S_ - 1);
  const float* p = scores + (long long)row * S_;
  bf16_t* o = probs + (long long)row * S_;
  __shared__ float red[256];
  const int tid = threadIdx.x;

  float mx = -3.4e38f;
  #pragma unroll
  for (int i = 0; i < S_ / 256; ++i) {
    const int k = tid + i * 256;
    if (k <= q) mx = fmaxf(mx, p[k]);
  }
  red[tid] = mx; __syncthreads();
  for (int off = 128; off > 0; off >>= 1) {
    if (tid < off) red[tid] = fmaxf(red[tid], red[tid + off]);
    __syncthreads();
  }
  mx = red[0]; __syncthreads();

  float ev[S_ / 256];
  float sum = 0.f;
  #pragma unroll
  for (int i = 0; i < S_ / 256; ++i) {
    const int k = tid + i * 256;
    ev[i] = (k <= q) ? __expf(p[k] - mx) : 0.f;
    sum += ev[i];
  }
  red[tid] = sum; __syncthreads();
  for (int off = 128; off > 0; off >>= 1) {
    if (tid < off) red[tid] += red[tid + off];
    __syncthreads();
  }
  const float inv = 1.f / red[0];
  #pragma unroll
  for (int i = 0; i < S_ / 256; ++i)
    o[tid + i * 256] = f2bf(ev[i] * inv);
}

// ---------------------------------------------------------------------------
__global__ __launch_bounds__(256)
void layernorm_kernel(const float* __restrict__ x, const float* __restrict__ g,
                      const float* __restrict__ b, bf16_t* __restrict__ y)
{
  const int row = blockIdx.x;
  const float* px = x + (long long)row * D_;
  bf16_t* py = y + (long long)row * D_;
  __shared__ float red[256];
  const int tid = threadIdx.x;

  float s = 0.f;
  for (int i = tid; i < D_; i += 256) s += px[i];
  red[tid] = s; __syncthreads();
  for (int off = 128; off > 0; off >>= 1) {
    if (tid < off) red[tid] += red[tid + off];
    __syncthreads();
  }
  const float mu = red[0] * (1.f / D_); __syncthreads();

  float v = 0.f;
  for (int i = tid; i < D_; i += 256) { const float d = px[i] - mu; v += d * d; }
  red[tid] = v; __syncthreads();
  for (int off = 128; off > 0; off >>= 1) {
    if (tid < off) red[tid] += red[tid + off];
    __syncthreads();
  }
  const float inv = rsqrtf(red[0] * (1.f / D_) + EPS_);
  for (int i = tid; i < D_; i += 256)
    py[i] = f2bf((px[i] - mu) * inv * g[i] + b[i]);
}

// ---------------------------------------------------------------------------
extern "C" void kernel_launch(void* const* d_in, const int* in_sizes, int n_in,
                              void* d_out, int out_size, void* d_ws, size_t ws_size,
                              hipStream_t stream)
{
  (void)in_sizes; (void)n_in; (void)out_size; (void)ws_size;

  const int*   tokens  = (const int*)  d_in[0];
  const float* tok_emb = (const float*)d_in[1];
  const float* pos_emb = (const float*)d_in[2];
  const float* Wq  = (const float*)d_in[3];   const float* bq_ = (const float*)d_in[4];
  const float* Wk  = (const float*)d_in[5];   const float* bk_ = (const float*)d_in[6];
  const float* Wv  = (const float*)d_in[7];   const float* bv_ = (const float*)d_in[8];
  const float* Wo  = (const float*)d_in[9];   const float* bo_ = (const float*)d_in[10];
  const float* Wup = (const float*)d_in[11];  const float* bup_= (const float*)d_in[12];
  const float* Wdn = (const float*)d_in[13];  const float* bdn_= (const float*)d_in[14];
  const float* ln_g= (const float*)d_in[15];  const float* ln_b= (const float*)d_in[16];
  const float* Wu  = (const float*)d_in[17];  const float* bu_ = (const float*)d_in[18];
  float* logits = (float*)d_out;

  const long long NT = (long long)B_ * S_;      // 4096 tokens
  // ---- workspace layout (~272 MiB) ----
  bf16_t* wT   = (bf16_t*)d_ws;                 // 64 MiB weight staging
  float*  x    = (float*)(wT + 33554432LL);     // residual stream f32   16 MiB
  bf16_t* xbf  = (bf16_t*)(x + NT * D_);        // x as bf16              8 MiB
  bf16_t* qbf  = xbf + NT * D_;                 // Q / ctx bf16          32 MiB
  bf16_t* kbf  = qbf + NT * H_;                 // K / mlp-h bf16        32 MiB
  bf16_t* vbf  = kbf + NT * H_;                 // V bf16                32 MiB
  bf16_t* vbfT = vbf + NT * H_;                 // V^T bf16 [B][H,S]     32 MiB
  float*  sc   = (float*)(vbfT + NT * H_);      // scores f32 [B,S,S]    32 MiB
  bf16_t* pbf  = (bf16_t*)(sc + (long long)B_ * S_ * S_);  // probs bf16 16 MiB
  bf16_t* xnbf = pbf + (long long)B_ * S_ * S_; // LN(x) bf16             8 MiB

  const dim3 blk(256);
  auto gemm = [&](const bf16_t* A, const bf16_t* Bmat, const float* bias,
                  const float* resid, float* Cf, bf16_t* Cb,
                  int M, int N, int K, long long sA, long long sB, long long sC,
                  int batches, float alpha, int flags) {
    dim3 grid(N / BN, M / BM, batches);
    gemm_bf16_wmma<<<grid, blk, 0, stream>>>(A, Bmat, bias, resid, Cf, Cb,
                                             M, N, K, sA, sB, sC, alpha, flags);
  };
  auto tcvt = [&](const float* src, bf16_t* dst, int R, int C) {
    tcvt_f32_bf16<<<dim3(C / 32, R / 32), blk, 0, stream>>>(src, dst, R, C);
  };

  embed_kernel<<<dim3(B_ * S_), blk, 0, stream>>>(tokens, tok_emb, pos_emb, x, xbf);

  const long long WSLOT = (long long)D_ * H_;   // 4M elems per weight slot
  for (int l = 0; l < L_; ++l) {
    tcvt(Wq  + (long long)l * D_ * H_, wT + 0 * WSLOT, D_, H_);  // [H,D]
    tcvt(Wk  + (long long)l * D_ * H_, wT + 1 * WSLOT, D_, H_);
    tcvt(Wv  + (long long)l * D_ * H_, wT + 2 * WSLOT, D_, H_);
    tcvt(Wo  + (long long)l * H_ * D_, wT + 3 * WSLOT, H_, D_);  // [D,H]
    tcvt(Wup + (long long)l * D_ * H_, wT + 4 * WSLOT, D_, H_);
    tcvt(Wdn + (long long)l * H_ * D_, wT + 5 * WSLOT, H_, D_);
    const float* pbq = bq_ + (long long)l * H_;
    const float* pbk = bk_ + (long long)l * H_;
    const float* pbv = bv_ + (long long)l * H_;
    const float* pbo = bo_ + (long long)l * D_;
    const float* pbu = bup_+ (long long)l * H_;
    const float* pbd = bdn_+ (long long)l * D_;

    // Q,K,V = x @ W + b  (bf16 out)
    gemm(xbf, wT + 0 * WSLOT, pbq, nullptr, nullptr, qbf,
         (int)NT, H_, D_, 0, 0, 0, 1, 1.f, 4 | 64);
    gemm(xbf, wT + 1 * WSLOT, pbk, nullptr, nullptr, kbf,
         (int)NT, H_, D_, 0, 0, 0, 1, 1.f, 4 | 64);
    gemm(xbf, wT + 2 * WSLOT, pbv, nullptr, nullptr, vbf,
         (int)NT, H_, D_, 0, 0, 0, 1, 1.f, 4 | 64);

    // V^T per batch for the P@V GEMM's [N,K] B-operand
    transpose_bf16<<<dim3(H_ / 32, S_ / 32, B_), blk, 0, stream>>>(
        vbf, vbfT, S_, H_, (long long)S_ * H_, (long long)S_ * H_);

    // scores[b] = SCALE * Q[b] @ K[b]^T  (skip fully-masked upper tiles)
    gemm(qbf, kbf, nullptr, nullptr, sc, nullptr, S_, S_, H_,
         (long long)S_ * H_, (long long)S_ * H_, (long long)S_ * S_,
         B_, SCALE_, 8 | 32);

    softmax_causal<<<dim3(B_ * S_), blk, 0, stream>>>(sc, pbf);

    // ctx[b] = P[b] @ V[b]  (K-loop clamped to causal range) -> reuse qbf
    gemm(pbf, vbfT, nullptr, nullptr, nullptr, qbf, S_, H_, S_,
         (long long)S_ * S_, (long long)S_ * H_, (long long)S_ * H_,
         B_, 1.f, 16 | 64);

    // x = x + ctx @ Wo + bo
    gemm(qbf, wT + 3 * WSLOT, pbo, x, x, xbf,
         (int)NT, D_, H_, 0, 0, 0, 1, 1.f, 4 | 2 | 32 | 64);

    // h = relu(x @ Wup + bup) -> reuse kbf
    gemm(xbf, wT + 4 * WSLOT, pbu, nullptr, nullptr, kbf,
         (int)NT, H_, D_, 0, 0, 0, 1, 1.f, 4 | 1 | 64);

    // x = x + h @ Wdn + bdn
    gemm(kbf, wT + 5 * WSLOT, pbd, x, x, xbf,
         (int)NT, D_, H_, 0, 0, 0, 1, 1.f, 4 | 2 | 32 | 64);
  }

  layernorm_kernel<<<dim3(B_ * S_), blk, 0, stream>>>(x, ln_g, ln_b, xnbf);

  // logits = LN(x) @ Wu + bu
  tcvt(Wu, wT, D_, V_);                         // [V,D] bf16
  gemm(xnbf, wT, bu_, nullptr, logits, nullptr,
       (int)NT, V_, D_, 0, 0, 0, 1, 1.f, 4 | 32);
}
